// LSTM_31018253812286
// MI455X (gfx1250) — compile-verified, matrix operand order checked by reading, not run on hardware
//
#include <hip/hip_runtime.h>

// ---------------------------------------------------------------------------
// Fused LSTM (T=5, H=256, I=13) + output projection for MI455X (gfx1250).
// One wave owns a 16-row batch tile; c-state and h live in registers/wave-
// private LDS for all 5 steps. W_hh chunks are double-buffered through LDS
// with gfx1250 async global->LDS copies. All matmuls use v_wmma_f32_16x16x32_f16
// with inline-0 C for the leading tile (bias folded into the pointwise stage).
// ---------------------------------------------------------------------------

typedef _Float16 v8h  __attribute__((ext_vector_type(8)));
typedef _Float16 v16h __attribute__((ext_vector_type(16)));
typedef float    v8f  __attribute__((ext_vector_type(8)));
typedef int      v4i  __attribute__((ext_vector_type(4)));

// ---- workspace layout (bytes) ----
#define WS_WHH   0         // 8 chunks * 128*256 f16 = 524288
#define WS_WIH   524288    // 8 chunks * 128*32  f16 = 65536
#define WS_W1    589824    // 16*1280 f16            = 40960
#define WS_BIAS  630784    // 8 chunks * 128 f32     = 4096

// ---- LDS layout (bytes) ----
#define LWHH_ROW  528      // 256 f16 + 8 f16 pad (16B aligned, bank-friendly)
#define LWIH_ROW  80       // 32 f16 + 8 f16 pad
#define LXS_ROW   80
#define LH_ROW    528
#define LWIH_OFF  67584    // 128*528
#define LBIAS_OFF 77824    // + 128*80
#define WBUF_SZ   78336
#define HST_OFF   156672   // 2*WBUF_SZ
#define HST_WAVE  8448     // 16*528
#define XST_OFF   224256   // + 8*8448
#define XST_WAVE  6400     // 5*16*80
#define SMEM_BYTES 275456  // + 8*6400  (~269 KB < 320 KB/WGP)

#if defined(__has_builtin)
#  if __has_builtin(__builtin_amdgcn_global_load_async_to_lds_b128)
#    define HAVE_ASYNC 1
#  endif
#  if __has_builtin(__builtin_amdgcn_s_wait_asynccnt)
#    define HAVE_WAITASYNC 1
#  endif
#endif
#ifndef HAVE_ASYNC
#  define HAVE_ASYNC 0
#endif

#if HAVE_ASYNC
typedef __attribute__((address_space(1))) v4i* gas_v4i_p;  // global
typedef __attribute__((address_space(3))) v4i* las_v4i_p;  // LDS
#endif

__device__ __forceinline__ void cp16(char* l, const char* g) {
#if HAVE_ASYNC
  __builtin_amdgcn_global_load_async_to_lds_b128((gas_v4i_p)g, (las_v4i_p)l, 0, 0);
#else
  *(v4i*)l = *(const v4i*)g;
#endif
}

__device__ __forceinline__ void wait_async0() {
#if HAVE_ASYNC
#  if defined(HAVE_WAITASYNC)
  __builtin_amdgcn_s_wait_asynccnt(0);
#  else
  asm volatile("s_wait_asynccnt 0" ::: "memory");
#  endif
#endif
}

__device__ __forceinline__ v16h ld_tile(const char* p) {
  v8h lo = *(const v8h*)p;
  v8h hi = *(const v8h*)(p + 16);
  return __builtin_shufflevector(lo, hi, 0,1,2,3,4,5,6,7,8,9,10,11,12,13,14,15);
}

__device__ __forceinline__ v8f wmma_f16(v16h a, v16h b, v8f c) {
  return __builtin_amdgcn_wmma_f32_16x16x32_f16(false, a, false, b, (short)0, c,
                                                false, false);
}

__device__ __forceinline__ float sigm(float x) {
  return __builtin_amdgcn_rcpf(1.0f + __builtin_amdgcn_exp2f(-1.44269504f * x));
}
__device__ __forceinline__ float tanhx(float x) {
  x = fminf(fmaxf(x, -10.0f), 10.0f);
  float e = __builtin_amdgcn_exp2f(2.88539008f * x);
  return (e - 1.0f) * __builtin_amdgcn_rcpf(e + 1.0f);
}

// Cooperative double-buffered stage of one gate chunk (W_hh 128x256 f16,
// W_ih 128x32 f16, fused bias 128 f32) into LDS.
__device__ __forceinline__ void issue_chunk(char* sm, const char* wsc, int cc,
                                            int b, int tid) {
  char* buf = sm + (b ? WBUF_SZ : 0);
  const char* g0 = wsc + WS_WHH + cc * 65536;
#pragma unroll
  for (int p = 0; p < 16; ++p) {
    int q = tid + 256 * p;
    cp16(buf + (q >> 5) * LWHH_ROW + (q & 31) * 16, g0 + q * 16);
  }
  const char* g1 = wsc + WS_WIH + cc * 8192;
  char* l1 = buf + LWIH_OFF;
#pragma unroll
  for (int p = 0; p < 2; ++p) {
    int q = tid + 256 * p;
    cp16(l1 + (q >> 2) * LWIH_ROW + (q & 3) * 16, g1 + q * 16);
  }
  if (tid < 32) {
    const char* g2 = wsc + WS_BIAS + cc * 512;
    cp16(buf + LBIAS_OFF + tid * 16, g2 + tid * 16);
  }
}

// ---------------------------------------------------------------------------
// Prep: convert weights to f16 chunk-major layout in workspace.
// Chunk c covers hidden cols [32c,32c+32); rows lr = g*32 + jj  ->  R = g*256+32c+jj
// ---------------------------------------------------------------------------
__global__ __launch_bounds__(256) void prep_kernel(
    const float* __restrict__ W_ih, const float* __restrict__ W_hh,
    const float* __restrict__ b_ih, const float* __restrict__ b_hh,
    const float* __restrict__ W1, char* __restrict__ ws) {
  int idx = blockIdx.x * 256 + threadIdx.x;
  _Float16* h = (_Float16*)ws;
  if (idx < 262144) {                       // W_hh chunks
    int c = idx >> 15, rem = idx & 32767;
    int lr = rem >> 8, k = rem & 255;
    int R = (lr >> 5) * 256 + c * 32 + (lr & 31);
    h[idx] = (_Float16)W_hh[R * 256 + k];
  } else if (idx < 294912) {                // W_ih chunks, K padded 13->32
    int i2 = idx - 262144;
    int c = i2 >> 12, rem = i2 & 4095;
    int lr = rem >> 5, i = rem & 31;
    int R = (lr >> 5) * 256 + c * 32 + (lr & 31);
    h[idx] = (i < 13) ? (_Float16)W_ih[R * 13 + i] : (_Float16)0.0f;
  } else if (idx < 315392) {                // W1 padded 6->16 rows
    int i3 = idx - 294912;
    int n = i3 / 1280, k = i3 - n * 1280;
    h[idx] = (n < 6) ? (_Float16)W1[n * 1280 + k] : (_Float16)0.0f;
  } else if (idx < 316416) {                // fused bias (f32)
    int i4 = idx - 315392;
    int c = i4 >> 7, lr = i4 & 127;
    int R = (lr >> 5) * 256 + c * 32 + (lr & 31);
    ((float*)(ws + WS_BIAS))[i4] = b_ih[R] + b_hh[R];
  }
}

// ---------------------------------------------------------------------------
// Main fused kernel: 256 threads = 8 waves; wave w owns batch rows
// [blk*128 + w*16, +16). 1024 blocks for B=131072.
// ---------------------------------------------------------------------------
__global__ __launch_bounds__(256) void lstm_main_kernel(
    const float* __restrict__ x, const float* __restrict__ b1,
    const char* __restrict__ wsc, float* __restrict__ out, int B) {
  extern __shared__ char smem[];
  const int tid = threadIdx.x;
  const int lane = tid & 31;
  const int wv = tid >> 5;
  const int laneN = lane & 15;
  const bool hiHalf = lane >= 16;
  const int row0 = blockIdx.x * 128 + wv * 16;

  char* hst = smem + HST_OFF + wv * HST_WAVE;  // wave-private h_t staging (f16)
  char* xst = smem + XST_OFF + wv * XST_WAVE;  // wave-private x staging (f16)

  // ---- stage x tile: f32 -> f16, zero-pad K 13->32 (wave-local, no barrier) ----
  {
    v4i z = {0, 0, 0, 0};
    for (int q = lane; q < 400; q += 32) *(v4i*)(xst + q * 16) = z;
    for (int e = lane; e < 1040; e += 32) {  // 16 rows * 5 t * 13 in
      int m = e / 65, rem = e - m * 65;
      int t = rem / 13, i = rem - t * 13;
      int r = row0 + m; if (r >= B) r = B - 1;
      float v = x[(size_t)r * 65 + rem];
      *(_Float16*)(xst + (t * 16 + m) * LXS_ROW + i * 2) = (_Float16)v;
    }
  }

  // ---- persistent per-wave state ----
  v8f cst[16];   // c-state: 16 D-tiles (16x256 f32)
  v16h hA[8];    // h_{t-1} in WMMA A layout (16x256 f16)
  v8f outacc;    // output accumulator (cols 0..5 valid)
  {
    v8f zf = {};
#pragma unroll
    for (int i = 0; i < 16; ++i) cst[i] = zf;
    v16h zh = {};
#pragma unroll
    for (int i = 0; i < 8; ++i) hA[i] = zh;
    float bv = (laneN < 6) ? b1[laneN] : 0.0f;
    v8f ob = {bv, bv, bv, bv, bv, bv, bv, bv};
    outacc = ob;
  }

  issue_chunk(smem, wsc, 0, 0, tid);  // prologue: chunk 0 -> buf 0
  const char* w1b = wsc + WS_W1;

#pragma unroll 1
  for (int t = 0; t < 5; ++t) {
    v16h xA = ld_tile(xst + t * (16 * LXS_ROW) + laneN * LXS_ROW + (hiHalf ? 32 : 0));

#pragma unroll
    for (int c = 0; c < 8; ++c) {
      wait_async0();
      __syncthreads();
      if (c < 7)          issue_chunk(smem, wsc, c + 1, (c + 1) & 1, tid);
      else if (t < 4)     issue_chunk(smem, wsc, 0, 0, tid);

      const char* wb = smem + ((c & 1) ? WBUF_SZ : 0);
      const char* lwih = wb + LWIH_OFF;
      const char* lbias = wb + LBIAS_OFF;

      // gates[16 x (4 gates x 32 cols)] = x_t@W_ih^T + h_{t-1}@W_hh^T
      // (C of the leading WMMA is the inline-0 operand; bias added pointwise)
      v8f acc[8];
#pragma unroll
      for (int g = 0; g < 4; ++g) {
#pragma unroll
        for (int u = 0; u < 2; ++u) {
          const int lr = g * 32 + u * 16 + laneN;
          v8f a = {};
          a = wmma_f16(xA, ld_tile(lwih + lr * LWIH_ROW + (hiHalf ? 32 : 0)), a);
#pragma unroll
          for (int kt = 0; kt < 8; ++kt)
            a = wmma_f16(hA[kt],
                         ld_tile(wb + lr * LWHH_ROW + kt * 64 + (hiHalf ? 32 : 0)),
                         a);
          acc[g * 2 + u] = a;
        }
      }

      // pointwise LSTM cell update (+bias); stage h_t chunk (f16) wave-locally
#pragma unroll
      for (int u = 0; u < 2; ++u) {
        const int j = c * 32 + u * 16 + laneN;
        const int lb = u * 16 + laneN;
        const float bI = *(const float*)(lbias + (0 * 32 + lb) * 4);
        const float bF = *(const float*)(lbias + (1 * 32 + lb) * 4);
        const float bG = *(const float*)(lbias + (2 * 32 + lb) * 4);
        const float bO = *(const float*)(lbias + (3 * 32 + lb) * 4);
#pragma unroll
        for (int r = 0; r < 8; ++r) {
          float ig = sigm(acc[0 + u][r] + bI);
          float fg = sigm(acc[2 + u][r] + bF);
          float gg = tanhx(acc[4 + u][r] + bG);
          float og = sigm(acc[6 + u][r] + bO);
          float cn = fg * cst[c * 2 + u][r] + ig * gg;
          cst[c * 2 + u][r] = cn;
          float hv = og * tanhx(cn);
          int m = r + (hiHalf ? 8 : 0);
          *(_Float16*)(hst + m * LH_ROW + j * 2) = (_Float16)hv;
        }
      }
    }

    // reload h_t in A layout: used by out-projection AND next step recurrence
#pragma unroll
    for (int kt = 0; kt < 8; ++kt)
      hA[kt] = ld_tile(hst + laneN * LH_ROW + kt * 64 + (hiHalf ? 32 : 0));

    // on-the-fly output projection: outacc += h_t @ W1[:, t*256:(t+1)*256]^T
#pragma unroll
    for (int kt = 0; kt < 8; ++kt)
      outacc = wmma_f16(
          hA[kt],
          ld_tile(w1b + laneN * 2560 + t * 512 + kt * 64 + (hiHalf ? 32 : 0)),
          outacc);
  }

  if (laneN < 6) {
#pragma unroll
    for (int r = 0; r < 8; ++r) {
      int m = r + (hiHalf ? 8 : 0);
      int row = row0 + m;
      if (row < B) out[(size_t)row * 6 + laneN] = outacc[r];
    }
  }
}

// ---------------------------------------------------------------------------
extern "C" void kernel_launch(void* const* d_in, const int* in_sizes, int n_in,
                              void* d_out, int out_size, void* d_ws,
                              size_t ws_size, hipStream_t stream) {
  const float* x    = (const float*)d_in[0];
  const float* W_ih = (const float*)d_in[1];
  const float* W_hh = (const float*)d_in[2];
  const float* b_ih = (const float*)d_in[3];
  const float* b_hh = (const float*)d_in[4];
  const float* W1   = (const float*)d_in[5];
  const float* b1   = (const float*)d_in[6];
  float* out = (float*)d_out;
  char* ws = (char*)d_ws;

  const int B = in_sizes[0] / 65;  // [B,5,13]

  prep_kernel<<<1236, 256, 0, stream>>>(W_ih, W_hh, b_ih, b_hh, W1, ws);

  const int nblk = (B + 127) / 128;
  lstm_main_kernel<<<nblk, 256, SMEM_BYTES, stream>>>(x, b1, ws, out, B);
}
